// AttentionBlock_16733192585778
// MI455X (gfx1250) — matrix operand order repaired and checked by reading, not compiled, not run on hardware
//
#include <hip/hip_runtime.h>
#include <hip/hip_bf16.h>

// ---------------------------------------------------------------------------
// AttentionBlock for MI455X (gfx1250, wave32, WMMA + Tensor Data Mover).
// B=32, C=512, H=W=32 -> N=1024 tokens. GROUPS=8.
// Pipeline:
//   1) gn_stats     : per-(b,g) mean/rstd
//   2) gn_norm      : xn(f32)->d_out (residual base) + h(f16)[b,n,c] (LDS transpose)
//   3) gemm_nt x3   : Q,K (mode0,+bias), V stored transposed (mode1) -> vT[b,d,m]
//   4) gemm_nt      : S = scale * Q*K^T  (f16, lives in L2: 64MB < 192MB)
//   5) softmax_rows : in-place over 1024 cols (f32 math)
//   6) gemm_nt      : O = P * vT^T
//   7) gemm_nt mode2: d_out = O*Wp^T + bp + xn (f32 residual epilogue)
// GEMMs use v_wmma_f32_16x16x32_f16; tiles staged by TENSOR_LOAD_TO_LDS (TDM)
// with double-buffered LDS and TENSORcnt pipelining.
// ---------------------------------------------------------------------------

typedef __attribute__((ext_vector_type(16))) _Float16 v16h;
typedef __attribute__((ext_vector_type(8)))  _Float16 v8h;
typedef __attribute__((ext_vector_type(8)))  float    v8f;
typedef __attribute__((ext_vector_type(4)))  unsigned int u32x4;
typedef __attribute__((ext_vector_type(8)))  int      i32x8;
typedef __attribute__((ext_vector_type(4)))  int      i32x4;

#define BATCH 32
#define CCH   512
#define NTOK  1024
#define GRPS  8

#if defined(__has_builtin)
#if __has_builtin(__builtin_amdgcn_tensor_load_to_lds) && __has_builtin(__builtin_amdgcn_s_wait_tensorcnt)
#define USE_TDM 1
#endif
#endif
#ifndef USE_TDM
#define USE_TDM 0
#endif

#if USE_TDM
// ---- Tensor Data Mover: 2D tile (tile_w elems x tile_h rows) global -> LDS ----
// D# packing per CDNA5 ISA.
// data_size code 1 = 2 bytes (f16). Groups 2/3 zero (2D tensor, VADDR2/3 unused).
__device__ inline void tdm_load_tile_f16(unsigned lds_byte_addr, const _Float16* gsrc,
                                         int tile_w, int tile_h, int stride_elems)
{
    const unsigned long long ga = (unsigned long long)(const void*)gsrc;
    const int tensor_w = stride_elems;      // >= tile_w; never read OOB here
    const int tensor_h = 1 << 20;

    u32x4 g0;
    g0[0] = 1u;                                              // count=1, user mode
    g0[1] = lds_byte_addr;                                   // lds_addr[31:0]
    g0[2] = (unsigned)(ga & 0xffffffffull);                  // global_addr[31:0]
    g0[3] = (unsigned)((ga >> 32) & 0x01ffffffull) | 0x80000000u; // addr[56:32] | type=2

    i32x8 g1;
    g1[0] = (1 << 16);                                       // wg_mask=0 | data_size=2B
    g1[1] = (tensor_w & 0xffff) << 16;                       // tensor_dim0[15:0]
    g1[2] = ((tensor_w >> 16) & 0xffff) | ((tensor_h & 0xffff) << 16);
    g1[3] = ((tensor_h >> 16) & 0xffff) | ((tile_w & 0xffff) << 16); // | tile_dim0
    g1[4] = (tile_h & 0xffff);                               // tile_dim1 | tile_dim2=0
    g1[5] = stride_elems;                                    // tensor_dim0_stride[31:0]
    g1[6] = 0;
    g1[7] = 0;

    const i32x4 z4 = {0, 0, 0, 0};
#if __has_include(<hip/amd_detail/amd_gfx1250_TDM.h>)
    const i32x8 z8 = {0, 0, 0, 0, 0, 0, 0, 0};
    __builtin_amdgcn_tensor_load_to_lds(g0, g1, z4, z4, z8, 0);   // clang-23 form
#else
    __builtin_amdgcn_tensor_load_to_lds(g0, g1, z4, z4, 0);       // ROCm 7.2 form
#endif
}
#endif

// ---- fragment loaders (CDNA5 16-bit WMMA VGPR layouts, CDNA5 ISA) ----
__device__ inline v16h load_frag_a(const _Float16* tile, int lane) {
    union { v16h v; v8h h[2]; } u;
    const int row = lane & 15, hi = lane >> 4;
    const v8h* p = (const v8h*)(tile + row * 32);
    u.h[0] = p[hi];
    u.h[1] = p[2 + hi];
    return u.v;
}
__device__ inline v16h load_frag_b(const _Float16* tile, int lane) {
    union { v16h v; v8h h[2]; } u;
    const int col = lane & 15, hi = lane >> 4;
    const v8h* p = (const v8h*)(tile + col * 32);
    u.h[0] = p[hi];
    u.h[1] = p[2 + hi];
    return u.v;
}

// ---------------------------------------------------------------------------
// Generic NT GEMM:  Y[m,n] = alpha * sum_k A[m,k]*Bm[n,k]  (+ bias[n])
//   mode 0: Yh[m*Nout + n] ; mode 1: Yh[n*M + m] ; mode 2: Yf[n*M + m] += v
// 256 thr (8 waves), tile 128x64, BK=32; double-buffered LDS fed by TDM.
// ---------------------------------------------------------------------------
__global__ __launch_bounds__(256)
void gemm_nt(const _Float16* __restrict__ A, long long sA,
             const _Float16* __restrict__ Bm, long long sB,
             const float* __restrict__ bias, float alpha,
             int M, int Nout, int K,
             _Float16* __restrict__ Yh, float* __restrict__ Yf,
             long long sY, int mode)
{
    __shared__ __align__(16) _Float16 As[2][128 * 32];
    __shared__ __align__(16) _Float16 Bs[2][64 * 32];

    const int b   = blockIdx.z;
    const _Float16* Ab = A  + (size_t)b * sA;
    const _Float16* Bb = Bm + (size_t)b * sB;

    const int m0   = blockIdx.x * 128;
    const int n0   = blockIdx.y * 64;
    const int tid  = threadIdx.x;
    const int wave = tid >> 5;
    const int lane = tid & 31;
    const int hi   = lane >> 4;
    const int l15  = lane & 15;
    const int nIt  = K >> 5;

    v8f acc[4] = {};

#if USE_TDM
    // Low 32 bits of a generic LDS pointer are the workgroup LDS byte offset.
    const unsigned asOff0 = (unsigned)(unsigned long long)(const void*)&As[0][0];
    const unsigned asOff1 = (unsigned)(unsigned long long)(const void*)&As[1][0];
    const unsigned bsOff0 = (unsigned)(unsigned long long)(const void*)&Bs[0][0];
    const unsigned bsOff1 = (unsigned)(unsigned long long)(const void*)&Bs[1][0];

    if (tid == 0) {   // one TDM descriptor per tile; EXEC ignored by tensor ops
        tdm_load_tile_f16(asOff0, Ab + (size_t)m0 * K, 32, 128, K);
        tdm_load_tile_f16(bsOff0, Bb + (size_t)n0 * K, 32, 64, K);
    }
    for (int it = 0; it < nIt; ++it) {
        const int cur = it & 1;
        if (it + 1 < nIt) {
            if (tid == 0) {   // prefetch next tiles into the other buffer
                const int k1 = (it + 1) << 5;
                tdm_load_tile_f16(cur ? asOff0 : asOff1, Ab + (size_t)m0 * K + k1, 32, 128, K);
                tdm_load_tile_f16(cur ? bsOff0 : bsOff1, Bb + (size_t)n0 * K + k1, 32, 64, K);
            }
            __builtin_amdgcn_s_wait_tensorcnt(2);  // current buffer's 2 loads done (in-order)
        } else {
            __builtin_amdgcn_s_wait_tensorcnt(0);
        }
        __syncthreads();

        const v16h a = load_frag_a(&As[cur][0] + wave * 16 * 32, lane);
#pragma unroll
        for (int g = 0; g < 4; ++g) {
            const v16h bf = load_frag_b(&Bs[cur][0] + g * 16 * 32, lane);
            acc[g] = __builtin_amdgcn_wmma_f32_16x16x32_f16(
                false, a, false, bf, (short)0, acc[g], false, false);
        }
        __syncthreads();   // compute done -> safe to DMA over this buffer next iter
    }
#else
    // Fallback: cooperative b128 copies (round-1 path).
    for (int k0 = 0; k0 < K; k0 += 32) {
        {
            const int r = tid >> 1, c = (tid & 1) * 16;
            const uint4* src = (const uint4*)(Ab + (size_t)(m0 + r) * K + k0 + c);
            uint4* dst = (uint4*)(&As[0][0] + r * 32 + c);
            dst[0] = src[0]; dst[1] = src[1];
        }
        if (tid < 128) {
            const int r = tid >> 1, c = (tid & 1) * 16;
            const uint4* src = (const uint4*)(Bb + (size_t)(n0 + r) * K + k0 + c);
            uint4* dst = (uint4*)(&Bs[0][0] + r * 32 + c);
            dst[0] = src[0]; dst[1] = src[1];
        }
        __syncthreads();
        const v16h a = load_frag_a(&As[0][0] + wave * 16 * 32, lane);
#pragma unroll
        for (int g = 0; g < 4; ++g) {
            const v16h bf = load_frag_b(&Bs[0][0] + g * 16 * 32, lane);
            acc[g] = __builtin_amdgcn_wmma_f32_16x16x32_f16(
                false, a, false, bf, (short)0, acc[g], false, false);
        }
        __syncthreads();
    }
#endif

    // Epilogue.  C/D layout: element r -> M = r + 8*hi, N = l15.
#pragma unroll
    for (int g = 0; g < 4; ++g) {
        const int n = n0 + g * 16 + l15;
        const float bv = bias ? bias[n] : 0.0f;
#pragma unroll
        for (int r = 0; r < 8; ++r) {
            const int m = m0 + wave * 16 + r + 8 * hi;
            const float v = alpha * acc[g][r] + bv;
            if (mode == 0) {
                Yh[(size_t)b * sY + (size_t)m * Nout + n] = (_Float16)v;
            } else if (mode == 1) {
                Yh[(size_t)b * sY + (size_t)n * M + m] = (_Float16)v;
            } else {
                const size_t idx = (size_t)b * sY + (size_t)n * M + m;
                Yf[idx] = Yf[idx] + v;   // residual add (xn already in d_out)
            }
        }
    }
}

// ---------------------------------------------------------------------------
__global__ __launch_bounds__(256)
void gn_stats(const float* __restrict__ x, float* __restrict__ stats)
{
    const int bg  = blockIdx.x;                 // b*8+g
    const float* p = x + (size_t)bg * 64 * 1024;
    const int tid = threadIdx.x;

    float s = 0.f, ss = 0.f;
    for (int i = tid; i < 65536; i += 256) {
        const float v = p[i];
        s += v; ss += v * v;
    }
    __shared__ float r1[256], r2[256];
    r1[tid] = s; r2[tid] = ss;
    __syncthreads();
    for (int off = 128; off > 0; off >>= 1) {
        if (tid < off) { r1[tid] += r1[tid + off]; r2[tid] += r2[tid + off]; }
        __syncthreads();
    }
    if (tid == 0) {
        const float mean = r1[0] * (1.0f / 65536.0f);
        const float var  = r2[0] * (1.0f / 65536.0f) - mean * mean;
        stats[2 * bg]     = mean;
        stats[2 * bg + 1] = rsqrtf(var + 1e-5f);
    }
}

// ---------------------------------------------------------------------------
__global__ __launch_bounds__(256)
void gn_norm(const float* __restrict__ x, const float* __restrict__ stats,
             const float* __restrict__ gamma, const float* __restrict__ beta,
             float* __restrict__ xn_out, _Float16* __restrict__ h)
{
    __shared__ float tile[32][33];
    const int b  = blockIdx.z;
    const int c0 = blockIdx.y * 32;
    const int w0 = blockIdx.x * 32;
    const int tx = threadIdx.x, ty = threadIdx.y;

#pragma unroll
    for (int i = 0; i < 4; ++i) {
        const int c  = c0 + ty + i * 8;
        const int hw = w0 + tx;
        const size_t idx = ((size_t)b * CCH + c) * NTOK + hw;
        const int g = c >> 6;
        const float mean = stats[2 * (b * GRPS + g)];
        const float rstd = stats[2 * (b * GRPS + g) + 1];
        const float xn = (x[idx] - mean) * rstd * gamma[c] + beta[c];
        xn_out[idx] = xn;
        tile[ty + i * 8][tx] = xn;
    }
    __syncthreads();
#pragma unroll
    for (int i = 0; i < 4; ++i) {
        const int hw = w0 + ty + i * 8;
        const int c  = c0 + tx;
        h[((size_t)b * NTOK + hw) * CCH + c] = (_Float16)tile[tx][ty + i * 8];
    }
}

// ---------------------------------------------------------------------------
__global__ __launch_bounds__(256)
void softmax_rows(_Float16* __restrict__ S)
{
    _Float16* p = S + (size_t)blockIdx.x * NTOK;
    const int tid = threadIdx.x;
    __shared__ float red[256];

    float v[4];
    float mx = -3.0e38f;
#pragma unroll
    for (int i = 0; i < 4; ++i) {
        v[i] = (float)p[tid + i * 256];
        mx = fmaxf(mx, v[i]);
    }
    red[tid] = mx;
    __syncthreads();
    for (int off = 128; off > 0; off >>= 1) {
        if (tid < off) red[tid] = fmaxf(red[tid], red[tid + off]);
        __syncthreads();
    }
    const float m = red[0];
    __syncthreads();

    float s = 0.f;
#pragma unroll
    for (int i = 0; i < 4; ++i) { v[i] = __expf(v[i] - m); s += v[i]; }
    red[tid] = s;
    __syncthreads();
    for (int off = 128; off > 0; off >>= 1) {
        if (tid < off) red[tid] += red[tid + off];
        __syncthreads();
    }
    const float inv = 1.0f / red[0];
#pragma unroll
    for (int i = 0; i < 4; ++i) p[tid + i * 256] = (_Float16)(v[i] * inv);
}

// ---------------------------------------------------------------------------
extern "C" void kernel_launch(void* const* d_in, const int* in_sizes, int n_in,
                              void* d_out, int out_size, void* d_ws, size_t ws_size,
                              hipStream_t stream)
{
    const float* x     = (const float*)d_in[0];
    const float* gamma = (const float*)d_in[1];
    const float* beta  = (const float*)d_in[2];
    const float* Wq = (const float*)d_in[3];
    const float* bq = (const float*)d_in[4];
    const float* Wk = (const float*)d_in[5];
    const float* bk = (const float*)d_in[6];
    const float* Wv = (const float*)d_in[7];
    const float* bv = (const float*)d_in[8];
    const float* Wp = (const float*)d_in[9];
    const float* bp = (const float*)d_in[10];
    (void)in_sizes; (void)n_in; (void)out_size; (void)ws_size;

    // Workspace: stats | h(32MB, aliased by O) | q | k | vT | S(64MB) | W16 x4
    char* ws = (char*)d_ws;
    float*    stats = (float*)ws;
    _Float16* h  = (_Float16*)(ws + 4096);
    const size_t HS = (size_t)BATCH * NTOK * CCH;
    _Float16* q  = h  + HS;
    _Float16* k  = q  + HS;
    _Float16* vT = k  + HS;
    _Float16* S  = vT + HS;
    const size_t SS = (size_t)BATCH * NTOK * NTOK;
    _Float16* w16 = S + SS;
    _Float16* Wq16 = w16, *Wk16 = w16 + 262144, *Wv16 = w16 + 524288, *Wp16 = w16 + 786432;
    _Float16* O  = h;

    {
        struct Cvt {
            static __global__ void run(const float* a, const float* b,
                                       const float* c, const float* d,
                                       _Float16* o) {
                const int i = blockIdx.x * 256 + threadIdx.x;
                o[i]          = (_Float16)a[i];
                o[i + 262144] = (_Float16)b[i];
                o[i + 524288] = (_Float16)c[i];
                o[i + 786432] = (_Float16)d[i];
            }
        };
        Cvt::run<<<dim3(1024), dim3(256), 0, stream>>>(Wq, Wk, Wv, Wp, w16);
    }

    gn_stats<<<dim3(BATCH * GRPS), dim3(256), 0, stream>>>(x, stats);
    gn_norm<<<dim3(NTOK / 32, CCH / 32, BATCH), dim3(32, 8), 0, stream>>>(
        x, stats, gamma, beta, (float*)d_out, h);

    const long long sHC = (long long)NTOK * CCH;   // 524288
    const long long sNN = (long long)NTOK * NTOK;  // 1048576

    gemm_nt<<<dim3(8, 8, BATCH), 256, 0, stream>>>(h, sHC, Wq16, 0, bq, 1.0f,
        NTOK, CCH, CCH, q, nullptr, sHC, 0);
    gemm_nt<<<dim3(8, 8, BATCH), 256, 0, stream>>>(h, sHC, Wk16, 0, bk, 1.0f,
        NTOK, CCH, CCH, k, nullptr, sHC, 0);
    gemm_nt<<<dim3(8, 8, BATCH), 256, 0, stream>>>(h, sHC, Wv16, 0, bv, 1.0f,
        NTOK, CCH, CCH, vT, nullptr, sHC, 1);

    gemm_nt<<<dim3(8, 16, BATCH), 256, 0, stream>>>(q, sHC, k, sHC, nullptr,
        0.044194173824159216f, NTOK, NTOK, CCH, S, nullptr, sNN, 0);

    softmax_rows<<<dim3(BATCH * NTOK), dim3(256), 0, stream>>>(S);

    gemm_nt<<<dim3(8, 8, BATCH), 256, 0, stream>>>(S, sNN, vT, sHC, nullptr, 1.0f,
        NTOK, CCH, NTOK, O, nullptr, sHC, 0);

    gemm_nt<<<dim3(8, 8, BATCH), 256, 0, stream>>>(O, sHC, Wp16, 0, bp, 1.0f,
        NTOK, CCH, CCH, nullptr, (float*)d_out, sHC, 2);
}
